// Half_Graph_28724741276432
// MI455X (gfx1250) — matrix operand order repaired and checked by reading, not compiled, not run on hardware
//
#include <hip/hip_runtime.h>
#include <hip/hip_bf16.h>
#include <math.h>

// ---------------- problem constants ----------------
#define Hc   96
#define Wc   96
#define NBc  4
#define Pc   (Hc*Wc)        // 9216 pixels; 96/16=6 -> a 16-pixel tile never straddles a row
#define HIDc 10
#define Sc   ((long)NBc*HIDc*Pc)   // 368640 elems per output tensor

typedef __attribute__((ext_vector_type(16))) _Float16 v16h;
typedef __attribute__((ext_vector_type(8)))  float    v8f;

// =====================================================================
// small pack / convert kernels
// =====================================================================
struct Segs {
    const float* p[5];
    long bs[5];   // per-batch element stride of each source
    int  ch[5];
    int  n;
};

// build f32->f16 channel-concatenated NCHW image: dst[b][Ctot][P]
__global__ void pack_concat_f16(Segs s, _Float16* __restrict__ dst, int Ctot) {
    int pix = blockIdx.x * 256 + threadIdx.x;
    if (pix >= Pc) return;
    int c = blockIdx.y, b = blockIdx.z;
    int seg = 0, cc = c;
    while (seg < s.n - 1 && cc >= s.ch[seg]) { cc -= s.ch[seg]; ++seg; }
    float v = s.p[seg][s.bs[seg] * b + (long)cc * Pc + pix];
    dst[((long)b * Ctot + c) * Pc + pix] = (_Float16)v;
}

// pack conv weight (Cout,Cin,kh,kw) -> f16 B image [Kpad][Npad], K-order c*KK+t
__global__ void pack_weights_f16(_Float16* __restrict__ dst, const float* __restrict__ w,
                                 int Cout, int Cin, int KK, int Kpad, int Npad) {
    int idx = blockIdx.x * 256 + threadIdx.x;
    if (idx >= Kpad * Npad) return;
    int k = idx / Npad, n = idx - (idx / Npad) * Npad;
    float v = 0.f;
    if (n < Cout && k < Cin * KK) {
        int c = k / KK, t = k - c * KK;
        v = w[((long)n * Cin + c) * KK + t];
    }
    dst[idx] = (_Float16)v;
}

__global__ void cvt_f16_kernel(const float* __restrict__ src, _Float16* __restrict__ dst, int n) {
    int i = blockIdx.x * 256 + threadIdx.x;
    if (i < n) dst[i] = (_Float16)src[i];
}

// =====================================================================
// WMMA fragment builders (ISA 7.12.2 layouts, wave32)
//   A 16x32 f16: lane = M row (lane&15); VGPR j holds K pair {2j,2j+1} (+8 hi-lanes, +16 for j>=4)
//   B 32x16 f16: lane = N col (lane&15); VGPR j holds K pair {2j,2j+1} (+8 hi-lanes, +16 for j>=4)
// =====================================================================
__device__ __forceinline__ v16h build_a_frag(const _Float16 sA[16][34], int lane) {
    v16h a;
    #pragma unroll
    for (int j = 0; j < 8; ++j) {
        int kb = (j < 4) ? 2 * j : 16 + 2 * (j - 4);
        if (lane >= 16) kb += 8;
        a[2 * j]     = sA[lane & 15][kb];
        a[2 * j + 1] = sA[lane & 15][kb + 1];
    }
    return a;
}
__device__ __forceinline__ v16h build_b_frag(const _Float16 sB[32][34], int lane, int nt) {
    v16h bf;
    #pragma unroll
    for (int j = 0; j < 8; ++j) {
        int kb = ((j < 4) ? 2 * j : 16 + 2 * (j - 4)) + ((lane >= 16) ? 8 : 0);
        bf[2 * j]     = sB[kb][nt * 16 + (lane & 15)];
        bf[2 * j + 1] = sB[kb + 1][nt * 16 + (lane & 15)];
    }
    return bf;
}

// =====================================================================
// generic implicit-GEMM conv (3x3 pad 1, or 1x1) with f16 WMMA, f32 accum
//   one wave32 per (16-pixel tile, batch); BOTH 16-col N tiles per wave
//   (A tile built once, 2 WMMAs per K-chunk -> halves staging for Cout>16)
//   mode: 0 = plain, 1 = +bias(p0), 2 = BN(g=p0,b=p1,m=p2,v=p3)+ReLU
// =====================================================================
__global__ __launch_bounds__(32)
void conv_wmma_kernel(const _Float16* __restrict__ img, int Cimg, int Cin, int KK,
                      const _Float16* __restrict__ Bw, int Kpad, int Npad, int Cout,
                      float* __restrict__ out, int mode,
                      const float* __restrict__ p0, const float* __restrict__ p1,
                      const float* __restrict__ p2, const float* __restrict__ p3) {
    __shared__ _Float16 sA[16][34];
    __shared__ _Float16 sB[32][34];
    const int lane = threadIdx.x;
    const int pix0 = blockIdx.x * 16;
    const int b    = blockIdx.y;
    const int y0   = pix0 / Wc;          // whole tile shares one row
    const int x0   = pix0 - y0 * Wc;
    const _Float16* imgb = img + (long)b * Cimg * Pc;
    const int  Ktot = Cin * KK;
    const bool twoN = (Npad > 16);
    const int  lgN  = twoN ? 5 : 4;
    v8f acc0 = {}, acc1 = {};

    for (int k0 = 0; k0 < Kpad; k0 += 32) {
        if (k0 + 32 < Kpad)   // prefetch next B panel chunk (global_prefetch_b8)
            __builtin_prefetch(Bw + (long)(k0 + 32) * Npad + lane * 2, 0, 1);
        // ---- stage A tile [16 pixels][32 K]; lanes sweep x (coalesced)
        #pragma unroll 4
        for (int e = 0; e < 16; ++e) {
            int idx = e * 32 + lane;
            int m = idx & 15, kk = idx >> 4;
            int kg = k0 + kk;
            float v = 0.f;
            if (kg < Ktot) {
                int c = kg / KK, t = kg - (kg / KK) * KK;
                int y = y0, x = x0 + m;
                if (KK == 9) { y += t / 3 - 1; x += t - (t / 3) * 3 - 1; }
                if (y >= 0 && y < Hc && x >= 0 && x < Wc)
                    v = (float)imgb[(long)c * Pc + y * Wc + x];
            }
            sA[m][kk] = (_Float16)v;
        }
        // ---- stage B tile [32 K][Npad N]
        for (int e = 0; e < Npad; ++e) {
            int idx = e * 32 + lane;
            int n = idx & (Npad - 1), kk = idx >> lgN;
            sB[kk][n] = Bw[(long)(k0 + kk) * Npad + n];
        }
        __syncthreads();
        v16h a = build_a_frag(sA, lane);
        v16h b0 = build_b_frag(sB, lane, 0);
        acc0 = __builtin_amdgcn_wmma_f32_16x16x32_f16(false, a, false, b0,
                                                      (short)0, acc0, false, false);
        if (twoN) {
            v16h b1 = build_b_frag(sB, lane, 1);
            acc1 = __builtin_amdgcn_wmma_f32_16x16x32_f16(false, a, false, b1,
                                                          (short)0, acc1, false, false);
        }
        __syncthreads();
    }
    // ---- epilogue (both N tiles)
    const int n = lane & 15;
    #pragma unroll
    for (int nt = 0; nt < 2; ++nt) {
        if (nt == 1 && !twoN) break;
        int o = nt * 16 + n;
        if (o >= Cout) continue;
        float sc = 1.f, sh = 0.f;
        if (mode == 1) sh = p0[o];
        else if (mode == 2) { sc = p0[o] * rsqrtf(p3[o] + 1e-5f); sh = p1[o] - p2[o] * sc; }
        float* ob = out + ((long)b * Cout + o) * Pc + pix0;
        const v8f& acc = nt ? acc1 : acc0;
        #pragma unroll
        for (int r = 0; r < 8; ++r) {
            int m = r + ((lane >= 16) ? 8 : 0);
            float y = acc[r] * sc + sh;
            if (mode == 2) y = fmaxf(y, 0.f);
            ob[m] = y;
        }
    }
}

// =====================================================================
// modulated deformable 3x3 conv: bilinear-sample einsum as WMMA GEMM.
//   off layout [b][27][P]: dy=off[2k], dx=off[2k+1], mask=sigmoid(off[18+k])
//   4 bilinear taps per (pixel,k) factored once, shared across all Cin channels
//   AND across both N tiles (dual accumulator). Epilogue: BN+ReLU.
// =====================================================================
__global__ __launch_bounds__(32)
void dfconv_wmma_kernel(const _Float16* __restrict__ img, int Cimg, int Cin,
                        const float* __restrict__ off,
                        const _Float16* __restrict__ Bw, int Kpad, int Npad, int Cout,
                        float* __restrict__ out,
                        const float* __restrict__ g,  const float* __restrict__ bt,
                        const float* __restrict__ mn, const float* __restrict__ vr) {
    __shared__ int      sIdx[144][4];
    __shared__ float    sWt[144][4];
    __shared__ _Float16 sA[16][34];
    __shared__ _Float16 sB[32][34];
    const int lane = threadIdx.x;
    const int pix0 = blockIdx.x * 16;
    const int b    = blockIdx.y;
    const int y0   = pix0 / Wc;
    const int x0   = pix0 - y0 * Wc;
    const float* offb = off + (long)b * 27 * Pc;

    // ---- precompute 16 pixels x 9 taps: 4 (index, weight*mask) pairs each
    for (int q = lane; q < 144; q += 32) {
        int m = q / 9, k = q - (q / 9) * 9;
        int pix = pix0 + m;
        float dy = offb[(2 * k) * Pc + pix];
        float dx = offb[(2 * k + 1) * Pc + pix];
        float msk = 1.f / (1.f + expf(-offb[(18 + k) * Pc + pix]));
        float ys = (float)y0 + (float)(k / 3 - 1) + dy;
        float xs = (float)(x0 + m) + (float)(k - (k / 3) * 3 - 1) + dx;
        float yf = floorf(ys), xf = floorf(xs);
        #pragma unroll
        for (int t = 0; t < 4; ++t) {
            float yc = yf + (float)(t >> 1), xc = xf + (float)(t & 1);
            float wgt = (1.f - fabsf(ys - yc)) * (1.f - fabsf(xs - xc));
            bool valid = (yc >= 0.f) && (yc <= (float)(Hc - 1)) &&
                         (xc >= 0.f) && (xc <= (float)(Wc - 1));
            int iy = (int)fminf(fmaxf(yc, 0.f), (float)(Hc - 1));
            int ix = (int)fminf(fmaxf(xc, 0.f), (float)(Wc - 1));
            sIdx[q][t] = iy * Wc + ix;
            sWt[q][t]  = valid ? wgt * msk : 0.f;
        }
    }
    __syncthreads();

    const _Float16* imgb = img + (long)b * Cimg * Pc;
    const int  Ktot = Cin * 9;
    const bool twoN = (Npad > 16);
    const int  lgN  = twoN ? 5 : 4;
    v8f acc0 = {}, acc1 = {};
    for (int k0 = 0; k0 < Kpad; k0 += 32) {
        if (k0 + 32 < Kpad)
            __builtin_prefetch(Bw + (long)(k0 + 32) * Npad + lane * 2, 0, 1);
        for (int e = 0; e < 16; ++e) {
            int idx = e * 32 + lane;
            int m = idx & 15, kk = idx >> 4;
            int kg = k0 + kk;
            float v = 0.f;
            if (kg < Ktot) {
                int c = kg / 9, k = kg - (kg / 9) * 9;
                int q = m * 9 + k;
                const _Float16* ic = imgb + (long)c * Pc;
                v = sWt[q][0] * (float)ic[sIdx[q][0]] + sWt[q][1] * (float)ic[sIdx[q][1]]
                  + sWt[q][2] * (float)ic[sIdx[q][2]] + sWt[q][3] * (float)ic[sIdx[q][3]];
            }
            sA[m][kk] = (_Float16)v;
        }
        for (int e = 0; e < Npad; ++e) {
            int idx = e * 32 + lane;
            int n = idx & (Npad - 1), kk = idx >> lgN;
            sB[kk][n] = Bw[(long)(k0 + kk) * Npad + n];
        }
        __syncthreads();
        v16h a = build_a_frag(sA, lane);
        v16h b0 = build_b_frag(sB, lane, 0);
        acc0 = __builtin_amdgcn_wmma_f32_16x16x32_f16(false, a, false, b0,
                                                      (short)0, acc0, false, false);
        if (twoN) {
            v16h b1 = build_b_frag(sB, lane, 1);
            acc1 = __builtin_amdgcn_wmma_f32_16x16x32_f16(false, a, false, b1,
                                                          (short)0, acc1, false, false);
        }
        __syncthreads();
    }
    const int n = lane & 15;
    #pragma unroll
    for (int nt = 0; nt < 2; ++nt) {
        if (nt == 1 && !twoN) break;
        int o = nt * 16 + n;
        if (o >= Cout) continue;
        float sc = g[o] * rsqrtf(vr[o] + 1e-5f);
        float sh = bt[o] - mn[o] * sc;
        float* ob = out + ((long)b * Cout + o) * Pc + pix0;
        const v8f& acc = nt ? acc1 : acc0;
        #pragma unroll
        for (int r = 0; r < 8; ++r) {
            int m = r + ((lane >= 16) ? 8 : 0);
            ob[m] = fmaxf(acc[r] * sc + sh, 0.f);
        }
    }
}

// =====================================================================
// per-pixel kernels (VALU; tiny FLOPs)
// =====================================================================
// decomp attention scalar: att = sigmoid(w2 . relu(bn(w1 . [xf,xh])) + b2)
__global__ __launch_bounds__(256)
void att_kernel(const float* __restrict__ xf, const float* __restrict__ xh,
                const float* __restrict__ w1, const float* __restrict__ g,
                const float* __restrict__ bt, const float* __restrict__ mn,
                const float* __restrict__ vr, const float* __restrict__ w2,
                const float* __restrict__ b2, float* __restrict__ att) {
    __shared__ float sw1[400], ssc[20], ssh[20], sw2[20];
    for (int i = threadIdx.x; i < 400; i += 256) sw1[i] = w1[i];
    for (int i = threadIdx.x; i < 20; i += 256) {
        float sc = g[i] * rsqrtf(vr[i] + 1e-5f);
        ssc[i] = sc; ssh[i] = bt[i] - mn[i] * sc; sw2[i] = w2[i];
    }
    __syncthreads();
    int idx = blockIdx.x * 256 + threadIdx.x;
    if (idx >= NBc * Pc) return;
    int b = idx / Pc, pix = idx - b * Pc;
    float in[20];
    for (int c = 0; c < 10; ++c) in[c]      = xf[((long)b * 10 + c) * Pc + pix];
    for (int c = 0; c < 10; ++c) in[10 + c] = xh[((long)b * 10 + c) * Pc + pix];
    float acc = b2[0];
    for (int o = 0; o < 20; ++o) {
        float s = 0.f;
        #pragma unroll
        for (int c = 0; c < 20; ++c) s += sw1[o * 20 + c] * in[c];
        acc += sw2[o] * fmaxf(s * ssc[o] + ssh[o], 0.f);
    }
    att[idx] = 1.f / (1.f + expf(-acc));
}

// msg = xph + xlh + relu(bn(att * t))   (fh_bn fused)
__global__ __launch_bounds__(256)
void msg_kernel(const float* __restrict__ xph, const float* __restrict__ xlh,
                const float* __restrict__ t,   const float* __restrict__ att,
                const float* __restrict__ g,   const float* __restrict__ bt,
                const float* __restrict__ mn,  const float* __restrict__ vr,
                float* __restrict__ msg) {
    int idx = blockIdx.x * 256 + threadIdx.x;
    if (idx >= NBc * HIDc * Pc) return;
    int b = idx / (HIDc * Pc);
    int r = idx - b * HIDc * Pc;
    int o = r / Pc, pix = r - o * Pc;
    float sc = g[o] * rsqrtf(vr[o] + 1e-5f);
    float xfh = fmaxf(att[(long)b * Pc + pix] * t[idx] * sc + (bt[o] - mn[o] * sc), 0.f);
    msg[idx] = xph[idx] + xlh[idx] + xfh;
}

// part_dep tail: out = (1 - sigmoid(Aw.pA+Ab)) * a * sigmoid(Bw.pB+Bb)
__global__ __launch_bounds__(256)
void pdep_kernel(const float* __restrict__ a, const float* __restrict__ pA,
                 const float* __restrict__ pB, const float* __restrict__ Aw,
                 const float* __restrict__ Ab, const float* __restrict__ Bw,
                 const float* __restrict__ Bb, float* __restrict__ out) {
    int idx = blockIdx.x * 256 + threadIdx.x;
    if (idx >= NBc * Pc) return;
    int b = idx / Pc, pix = idx - b * Pc;
    float sa = Ab[0], sb = Bb[0];
    for (int c = 0; c < 10; ++c) {
        sa += Aw[c] * pA[((long)b * 10 + c) * Pc + pix];
        sb += Bw[c] * pB[((long)b * 10 + c) * Pc + pix];
    }
    float fa = 1.f - 1.f / (1.f + expf(-sa));
    float fb = 1.f / (1.f + expf(-sb));
    for (int o = 0; o < 10; ++o) {
        long i = ((long)b * 10 + o) * Pc + pix;
        out[i] = fa * a[i] * fb;
    }
}

// 2-layer Conv2dGRU, seq_len=1, h0=0 => gh is just b_hh, h = (1-z)*n
__global__ __launch_bounds__(256)
void gru_kernel(const float* __restrict__ xh, const float* __restrict__ msg,
                const float* __restrict__ w0, const float* __restrict__ bih0,
                const float* __restrict__ bhh0, const float* __restrict__ w1,
                const float* __restrict__ bih1, const float* __restrict__ bhh1,
                float* __restrict__ out) {
    __shared__ float sw0[600], sw1[300], sb[120];
    for (int i = threadIdx.x; i < 600; i += 256) sw0[i] = w0[i];
    for (int i = threadIdx.x; i < 300; i += 256) sw1[i] = w1[i];
    for (int i = threadIdx.x; i < 30; i += 256) {
        sb[i] = bih0[i]; sb[30 + i] = bhh0[i]; sb[60 + i] = bih1[i]; sb[90 + i] = bhh1[i];
    }
    __syncthreads();
    int idx = blockIdx.x * 256 + threadIdx.x;
    if (idx >= NBc * Pc) return;
    int b = idx / Pc, pix = idx - b * Pc;
    float x[20], h[10];
    for (int c = 0; c < 10; ++c) x[c]      = xh[((long)b * 10 + c) * Pc + pix];
    for (int c = 0; c < 10; ++c) x[10 + c] = msg[((long)b * 10 + c) * Pc + pix];
    for (int o = 0; o < 10; ++o) {
        float gr = sb[o], gz = sb[10 + o], gn = sb[20 + o];
        #pragma unroll
        for (int c = 0; c < 20; ++c) {
            gr += sw0[o * 20 + c] * x[c];
            gz += sw0[(10 + o) * 20 + c] * x[c];
            gn += sw0[(20 + o) * 20 + c] * x[c];
        }
        float rr = 1.f / (1.f + expf(-(gr + sb[30 + o])));
        float zz = 1.f / (1.f + expf(-(gz + sb[40 + o])));
        float nn = tanhf(gn + rr * sb[50 + o]);
        h[o] = (1.f - zz) * nn;
    }
    for (int o = 0; o < 10; ++o) {
        float gr = sb[60 + o], gz = sb[70 + o], gn = sb[80 + o];
        #pragma unroll
        for (int c = 0; c < 10; ++c) {
            gr += sw1[o * 10 + c] * h[c];
            gz += sw1[(10 + o) * 10 + c] * h[c];
            gn += sw1[(20 + o) * 10 + c] * h[c];
        }
        float rr = 1.f / (1.f + expf(-(gr + sb[90 + o])));
        float zz = 1.f / (1.f + expf(-(gz + sb[100 + o])));
        float nn = tanhf(gn + rr * sb[110 + o]);
        out[((long)b * 10 + o) * Pc + pix] = (1.f - zz) * nn;
    }
}

// =====================================================================
// host orchestration
// =====================================================================
extern "C" void kernel_launch(void* const* d_in, const int* in_sizes, int n_in,
                              void* d_out, int out_size, void* d_ws, size_t ws_size,
                              hipStream_t stream) {
    // ---- input leaf indices (top level = setup_inputs insertion order,
    //      params = pytree flatten with sorted dict keys)
    const float* h_fea = (const float*)d_in[0];
    const float* xh0   = (const float*)d_in[1];
    const float* xh1   = (const float*)d_in[2];
    const float* xf    = (const float*)d_in[3];
    const float* xp    = (const float*)d_in[4];
    auto F = [&](int i) { return (const float*)d_in[i]; };
    // conv_phl 5..14, conv_phu 15..24, dec_l 25..36, dec_u 37..48,
    // dp_h 49..66, dp_l 67..84, gru_l 85..92, gru_u 93..100
    (void)in_sizes; (void)n_in; (void)out_size; (void)ws_size;

    // ---- workspace bump allocator
    char* wp = (char*)d_ws;
    auto alloc = [&](size_t bytes) -> void* {
        void* r = (void*)wp;
        wp += (bytes + 255) & ~(size_t)255;
        return r;
    };
    const long BP = (long)NBc * Pc;
    _Float16* in_dp_h  = (_Float16*)alloc(266 * BP * 2);
    _Float16* in_dp_l  = (_Float16*)alloc(266 * BP * 2);
    _Float16* in_phu   = (_Float16*)alloc(50 * BP * 2);
    _Float16* in_phl   = (_Float16*)alloc(30 * BP * 2);
    _Float16* phu_m16  = (_Float16*)alloc(20 * BP * 2);
    _Float16* phl_m16  = (_Float16*)alloc(20 * BP * 2);
    _Float16* a_h1_16  = (_Float16*)alloc(20 * BP * 2);
    _Float16* a_l1_16  = (_Float16*)alloc(20 * BP * 2);
    float* phu_mid = (float*)alloc(20 * BP * 4);
    float* phl_mid = (float*)alloc(20 * BP * 4);
    float* xphu    = (float*)alloc(10 * BP * 4);
    float* xphl    = (float*)alloc(10 * BP * 4);
    float* off_h   = (float*)alloc(27 * BP * 4);   // reused d1 then d2
    float* off_l   = (float*)alloc(27 * BP * 4);
    float* a_h1    = (float*)alloc(20 * BP * 4);
    float* a_l1    = (float*)alloc(20 * BP * 4);
    float* a_h2    = (float*)alloc(10 * BP * 4);
    float* a_l2    = (float*)alloc(10 * BP * 4);
    float* xlh     = (float*)alloc(10 * BP * 4);
    float* xuh     = (float*)alloc(10 * BP * 4);
    float* att_u   = (float*)alloc(1 * BP * 4);
    float* att_l   = (float*)alloc(1 * BP * 4);
    float* t_u     = (float*)alloc(10 * BP * 4);
    float* t_l     = (float*)alloc(10 * BP * 4);
    auto allocW = [&](int Kpad, int Npad) { return (_Float16*)alloc((size_t)Kpad * Npad * 2); };
    _Float16 *Bw_phu1 = allocW(480, 32),  *Bw_phu2 = allocW(192, 16);
    _Float16 *Bw_phl1 = allocW(288, 32),  *Bw_phl2 = allocW(192, 16);
    _Float16 *Bw_offh1 = allocW(2400, 32), *Bw_wh1 = allocW(2400, 32);
    _Float16 *Bw_offh2 = allocW(192, 32),  *Bw_wh2 = allocW(192, 16);
    _Float16 *Bw_offl1 = allocW(2400, 32), *Bw_wl1 = allocW(2400, 32);
    _Float16 *Bw_offl2 = allocW(192, 32),  *Bw_wl2 = allocW(192, 16);
    _Float16 *Bw_fhu = allocW(256, 16),    *Bw_fhl = allocW(256, 16);

    // ---- helpers
    auto packW = [&](_Float16* dst, const float* w, int Cout, int Cin, int KK,
                     int Kpad, int Npad) {
        int n = Kpad * Npad;
        pack_weights_f16<<<(n + 255) / 256, 256, 0, stream>>>(dst, w, Cout, Cin, KK, Kpad, Npad);
    };
    auto conv = [&](const _Float16* img, int Cimg, int Cin, int KK,
                    const _Float16* Bw, int Kpad, int Npad, int Cout, float* out,
                    int mode, const float* p0, const float* p1, const float* p2,
                    const float* p3) {
        dim3 gr(Pc / 16, NBc);
        conv_wmma_kernel<<<gr, 32, 0, stream>>>(img, Cimg, Cin, KK, Bw, Kpad, Npad,
                                                Cout, out, mode, p0, p1, p2, p3);
    };
    auto dfconv = [&](const _Float16* img, int Cimg, int Cin, const float* off,
                      const _Float16* Bw, int Kpad, int Npad, int Cout, float* out,
                      const float* g, const float* bt, const float* mn, const float* vr) {
        dim3 gr(Pc / 16, NBc);
        dfconv_wmma_kernel<<<gr, 32, 0, stream>>>(img, Cimg, Cin, off, Bw, Kpad, Npad,
                                                  Cout, out, g, bt, mn, vr);
    };
    auto cvt = [&](const float* s, _Float16* d, long n) {
        cvt_f16_kernel<<<(int)((n + 255) / 256), 256, 0, stream>>>(s, d, (int)n);
    };
    auto concat = [&](Segs s, _Float16* dst, int Ctot) {
        dim3 gr(Pc / 256, Ctot, NBc);
        pack_concat_f16<<<gr, 256, 0, stream>>>(s, dst, Ctot);
    };

    // ---- weight packs
    packW(Bw_phl1, F(13), 20, 30, 9, 288, 32);
    packW(Bw_phl2, F(14), 10, 20, 9, 192, 16);
    packW(Bw_phu1, F(23), 20, 50, 9, 480, 32);
    packW(Bw_phu2, F(24), 10, 20, 9, 192, 16);
    packW(Bw_offh1, F(62), 27, 266, 9, 2400, 32);
    packW(Bw_wh1,   F(63), 20, 266, 9, 2400, 32);
    packW(Bw_offh2, F(65), 27, 20, 9, 192, 32);
    packW(Bw_wh2,   F(66), 10, 20, 9, 192, 16);
    packW(Bw_offl1, F(80), 27, 266, 9, 2400, 32);
    packW(Bw_wl1,   F(81), 20, 266, 9, 2400, 32);
    packW(Bw_offl2, F(83), 27, 20, 9, 192, 32);
    packW(Bw_wl2,   F(84), 10, 20, 9, 192, 16);
    packW(Bw_fhu, F(48), 10, 256, 1, 256, 16);
    packW(Bw_fhl, F(36), 10, 256, 1, 256, 16);

    // ---- f16 concat images
    const long hfS = 256L * Pc, hS = 10L * Pc, xpS = (long)NBc * 10 * Pc;
    { Segs s{}; s.n = 2; s.p[0] = h_fea; s.ch[0] = 256; s.bs[0] = hfS;
      s.p[1] = xh1; s.ch[1] = 10; s.bs[1] = hS; concat(s, in_dp_h, 266); }
    { Segs s{}; s.n = 2; s.p[0] = h_fea; s.ch[0] = 256; s.bs[0] = hfS;
      s.p[1] = xh0; s.ch[1] = 10; s.bs[1] = hS; concat(s, in_dp_l, 266); }
    { Segs s{}; s.n = 5;
      for (int k = 0; k < 4; ++k) { s.p[k] = xp + k * xpS; s.ch[k] = 10; s.bs[k] = hS; }
      s.p[4] = xh0; s.ch[4] = 10; s.bs[4] = hS; concat(s, in_phu, 50); }
    { Segs s{}; s.n = 3;
      s.p[0] = xp + 4 * xpS; s.ch[0] = 10; s.bs[0] = hS;
      s.p[1] = xp + 5 * xpS; s.ch[1] = 10; s.bs[1] = hS;
      s.p[2] = xh1; s.ch[2] = 10; s.bs[2] = hS; concat(s, in_phl, 30); }

    // ---- conv_ph upper/lower
    conv(in_phu, 50, 50, 9, Bw_phu1, 480, 32, 20, phu_mid, 2, F(16), F(15), F(17), F(18));
    cvt(phu_mid, phu_m16, 20 * BP);
    conv(phu_m16, 20, 20, 9, Bw_phu2, 192, 16, 10, xphu, 2, F(20), F(19), F(21), F(22));
    conv(in_phl, 30, 30, 9, Bw_phl1, 288, 32, 20, phl_mid, 2, F(6), F(5), F(7), F(8));
    cvt(phl_mid, phl_m16, 20 * BP);
    conv(phl_m16, 20, 20, 9, Bw_phl2, 192, 16, 10, xphl, 2, F(10), F(9), F(11), F(12));

    // ---- part_dep dp_h (pA=xh1, pB=xh0)
    conv(in_dp_h, 266, 266, 9, Bw_offh1, 2400, 32, 27, off_h, 1, F(61), 0, 0, 0);
    dfconv(in_dp_h, 266, 266, off_h, Bw_wh1, 2400, 32, 20, a_h1, F(54), F(53), F(55), F(56));
    cvt(a_h1, a_h1_16, 20 * BP);
    conv(a_h1_16, 20, 20, 9, Bw_offh2, 192, 32, 27, off_h, 1, F(64), 0, 0, 0);
    dfconv(a_h1_16, 20, 20, off_h, Bw_wh2, 192, 16, 10, a_h2, F(58), F(57), F(59), F(60));
    pdep_kernel<<<(NBc * Pc + 255) / 256, 256, 0, stream>>>(a_h2, xh1, xh0,
        F(50), F(49), F(52), F(51), xlh);

    // ---- part_dep dp_l (pA=xh0, pB=xh1)
    conv(in_dp_l, 266, 266, 9, Bw_offl1, 2400, 32, 27, off_l, 1, F(79), 0, 0, 0);
    dfconv(in_dp_l, 266, 266, off_l, Bw_wl1, 2400, 32, 20, a_l1, F(72), F(71), F(73), F(74));
    cvt(a_l1, a_l1_16, 20 * BP);
    conv(a_l1_16, 20, 20, 9, Bw_offl2, 192, 32, 27, off_l, 1, F(82), 0, 0, 0);
    dfconv(a_l1_16, 20, 20, off_l, Bw_wl2, 192, 16, 10, a_l2, F(76), F(75), F(77), F(78));
    pdep_kernel<<<(NBc * Pc + 255) / 256, 256, 0, stream>>>(a_l2, xh0, xh1,
        F(68), F(67), F(70), F(69), xuh);

    // ---- decomp u/l: att scalar + (fh_w . h_fea), scaled per-pixel later
    att_kernel<<<(NBc * Pc + 255) / 256, 256, 0, stream>>>(xf, xh0,
        F(42), F(39), F(38), F(40), F(41), F(43), F(37), att_u);
    att_kernel<<<(NBc * Pc + 255) / 256, 256, 0, stream>>>(xf, xh1,
        F(30), F(27), F(26), F(28), F(29), F(31), F(25), att_l);
    conv(in_dp_h, 266, 256, 1, Bw_fhu, 256, 16, 10, t_u, 0, 0, 0, 0, 0);
    conv(in_dp_h, 266, 256, 1, Bw_fhl, 256, 16, 10, t_l, 0, 0, 0, 0, 0);

    // ---- messages (outputs 2 and 3) then GRUs (outputs 0 and 1)
    float* out = (float*)d_out;
    float* msg_u = out + 2 * Sc;
    float* msg_l = out + 3 * Sc;
    msg_kernel<<<(int)((10 * BP + 255) / 256), 256, 0, stream>>>(xphu, xlh, t_u, att_u,
        F(45), F(44), F(46), F(47), msg_u);
    msg_kernel<<<(int)((10 * BP + 255) / 256), 256, 0, stream>>>(xphl, xuh, t_l, att_l,
        F(33), F(32), F(34), F(35), msg_l);
    gru_kernel<<<(NBc * Pc + 255) / 256, 256, 0, stream>>>(xh0, msg_u,
        F(96), F(94), F(93), F(100), F(98), F(97), out);
    gru_kernel<<<(NBc * Pc + 255) / 256, 256, 0, stream>>>(xh1, msg_l,
        F(88), F(86), F(85), F(92), F(90), F(89), out + Sc);
}
